// MoE_18124761989478
// MI455X (gfx1250) — compile-verified, matrix operand order checked by reading, not run on hardware
//
#include <hip/hip_runtime.h>
#include <hip/hip_bf16.h>
#include <stdint.h>

// ---------------------------------------------------------------------------
// MoE (top-2 of 8 experts), D=1024, H=4096, T=8192 tokens.
// route -> gather per-expert token lists -> fused grouped FFN with bf16 WMMA.
// M=32 token tiles (2 stacked WMMA m-tiles) to halve L2 weight traffic;
// x-tile gathered via global_load_async_to_lds_b128 (ASYNCcnt DMA path).
// ---------------------------------------------------------------------------

#define T_TOK   8192                 // B*S
#define D_DIM   1024
#define H_DIM   4096
#define E_EXP   8
#define M_TILE  32                   // tokens per block (2 x WMMA M)
#define TILES   (T_TOK / M_TILE)     // 256
#define QH      1024                 // h columns resident per quarter
#define KSTAGE  64                   // K-depth staged per LDS weight panel
#define NCHUNK  128                  // panel columns (8 waves * 16)

typedef __bf16 bf16;
typedef bf16  v16bf __attribute__((ext_vector_type(16)));
typedef bf16  v8bf  __attribute__((ext_vector_type(8)));
typedef float v8f   __attribute__((ext_vector_type(8)));

static __device__ __forceinline__ float gelu_exact(float v) {
    return 0.5f * v * (1.0f + erff(v * 0.70710678118654752440f));
}

// 16-element bf16 fragment from two contiguous 16B LDS chunks (A/B striping:
// each lane holds two runs of 8 consecutive K values selected by lane[4]).
static __device__ __forceinline__ v16bf frag16(const bf16* lo, const bf16* hi) {
    v8bf a = *(const v8bf*)lo;
    v8bf b = *(const v8bf*)hi;
    return __builtin_shufflevector(a, b, 0, 1, 2, 3, 4, 5, 6, 7,
                                         8, 9, 10, 11, 12, 13, 14, 15);
}

// Async DMA: 16 bytes global -> LDS, tracked by ASYNCcnt.
static __device__ __forceinline__ void async_copy16(uint32_t lds_off,
                                                    const void* gptr) {
    unsigned long long ga = (unsigned long long)gptr;
    asm volatile("global_load_async_to_lds_b128 %0, %1, off"
                 :: "v"(lds_off), "v"(ga) : "memory");
}
static __device__ __forceinline__ void async_wait0() {
    asm volatile("s_wait_asynccnt 0x0" ::: "memory");
}

// ---------------------------------------------------------------------------
// Router: logits = x @ Wg + bg ; softmax over E=8 ; top-2 (torch semantics);
// scatter token ids + gate weights into per-expert lists.
// ---------------------------------------------------------------------------
__global__ __launch_bounds__(256) void moe_router(
    const float* __restrict__ x, const float* __restrict__ Wg,
    const float* __restrict__ bg, int* __restrict__ counts,
    int* __restrict__ tlist, float* __restrict__ wlist)
{
    const int t = blockIdx.x * blockDim.x + threadIdx.x;
    if (t >= T_TOK) return;

    float acc[E_EXP];
#pragma unroll
    for (int e = 0; e < E_EXP; ++e) acc[e] = bg[e];

    const float* xr = x + (size_t)t * D_DIM;
    for (int d = 0; d < D_DIM; ++d) {
        const float xv = xr[d];
        const float* wr = Wg + (size_t)d * E_EXP;
#pragma unroll
        for (int e = 0; e < E_EXP; ++e) acc[e] = fmaf(xv, wr[e], acc[e]);
    }

    float m = acc[0];
#pragma unroll
    for (int e = 1; e < E_EXP; ++e) m = fmaxf(m, acc[e]);
    float s = 0.f, p[E_EXP];
#pragma unroll
    for (int e = 0; e < E_EXP; ++e) { p[e] = __expf(acc[e] - m); s += p[e]; }
    const float inv = 1.0f / s;
#pragma unroll
    for (int e = 0; e < E_EXP; ++e) p[e] *= inv;

    int e1 = 0;
#pragma unroll
    for (int e = 1; e < E_EXP; ++e) if (p[e] > p[e1]) e1 = e;
    int e2 = (e1 == 0) ? 1 : 0;
#pragma unroll
    for (int e = 0; e < E_EXP; ++e)
        if (e != e1 && p[e] > p[e2]) e2 = e;

    int pos1 = atomicAdd(&counts[e1], 1);
    tlist[e1 * T_TOK + pos1] = t;
    wlist[e1 * T_TOK + pos1] = p[e1];
    int pos2 = atomicAdd(&counts[e2], 1);
    tlist[e2 * T_TOK + pos2] = t;
    wlist[e2 * T_TOK + pos2] = p[e2];
}

// ---------------------------------------------------------------------------
// Fused expert FFN: one block = (expert, 32-token tile), 256 thr = 8 wave32.
// Quarter-H pipeline: for q in 0..3:
//   hs[32][1024] = gelu(x @ W1[:, q])   (bf16, LDS)
//   yacc[dc][mt] += hs @ W2[q, :]       (f32 accum, 128 VGPRs persist)
// then bias + gate scale + f32 atomic scatter (2 commutative adds/element).
// Weight panels: global f32 -> cvt -> LDS bf16, KSTAGE x NCHUNK, prefetched.
// x gather: global_load_async_to_lds_b128 -> LDS f32 arena -> cvt bf16.
// ---------------------------------------------------------------------------
__global__ __launch_bounds__(256) void moe_expert_ffn(
    const float* __restrict__ x,
    const float* __restrict__ W1, const float* __restrict__ b1,
    const float* __restrict__ W2, const float* __restrict__ b2,
    const int* __restrict__ counts, const int* __restrict__ tlist,
    const float* __restrict__ wlist, float* __restrict__ out)
{
    const int e    = blockIdx.x >> 8;            // blockIdx.x / TILES
    const int tile = blockIdx.x & (TILES - 1);
    const int cnt  = counts[e];
    if (tile * M_TILE >= cnt) return;

    const int tid  = threadIdx.x;
    const int wave = tid >> 5;                   // wave32
    const int lane = tid & 31;
    const int lm   = lane & 15;
    const int ksel = lane >> 4;

    // +8 halfword pad keeps fragment bases 16B aligned; strides stay 16B mult.
    __shared__ bf16 xs[M_TILE][D_DIM + 8];       // x tile bf16      (~66 KB)
    __shared__ bf16 hs[M_TILE][QH + 8];          // h quarter bf16   (~66 KB)
    // Arena aliases: bf16 weight panel [NCHUNK][KSTAGE+8]  (18,432 B)
    //                f32 x staging     [4][1024]           (16,384 B)
    __shared__ __align__(16) char arena[NCHUNK * (KSTAGE + 8) * sizeof(bf16)];
    __shared__ int   tk[M_TILE];
    __shared__ float tw[M_TILE];
    bf16 (*Bls)[KSTAGE + 8] = (bf16(*)[KSTAGE + 8])arena;
    float* xstage = (float*)arena;

    if (tid < M_TILE) {
        int pos = tile * M_TILE + tid;
        int cl  = min(pos, cnt - 1);
        tk[tid] = tlist[e * T_TOK + cl];
        tw[tid] = (pos < cnt) ? wlist[e * T_TOK + cl] : 0.0f;
    }
    __syncthreads();

    // ---- Gather x tile: async DMA 4 rows of f32 into arena, cvt to bf16 ----
    for (int g = 0; g < M_TILE / 4; ++g) {
#pragma unroll
        for (int j = 0; j < 4; ++j) {                     // 4 x 16B per thread
            int c   = tid + j * 256;                      // chunk id 0..1023
            int r   = c >> 8;                             // row in group
            int col = (c & 255) * 4;                      // f32 column
            uint32_t dst = (uint32_t)(uintptr_t)&xstage[c * 4];
            async_copy16(dst, x + (size_t)tk[g * 4 + r] * D_DIM + col);
        }
        async_wait0();
        __syncthreads();
        for (int i = tid; i < 4 * D_DIM; i += 256) {
            int r = i >> 10, col = i & (D_DIM - 1);
            xs[g * 4 + r][col] = (bf16)xstage[i];
        }
        __syncthreads();
    }

    const float* W1g = W1 + (size_t)e * D_DIM * H_DIM;
    const float* W2g = W2 + (size_t)e * H_DIM * D_DIM;
    const float* b1g = b1 + (size_t)e * H_DIM;
    const float* b2g = b2 + (size_t)e * D_DIM;

    v8f yacc[8][2];
#pragma unroll
    for (int dc = 0; dc < 8; ++dc)
#pragma unroll
        for (int mt = 0; mt < 2; ++mt) yacc[dc][mt] = (v8f)(0.0f);

    for (int q = 0; q < H_DIM / QH; ++q) {
        // ------------- Phase 1: hs = gelu(x @ W1[:, q*QH .. +QH) + b1) -----
        for (int nc = 0; nc < QH / NCHUNK; ++nc) {          // 8 chunks
            const int gcol0 = q * QH + nc * NCHUNK;
            v8f acc[2];
            acc[0] = (v8f)(0.0f);
            acc[1] = (v8f)(0.0f);
            for (int ks = 0; ks < D_DIM; ks += KSTAGE) {    // 16 stages
                const float* src = W1g + (size_t)ks * H_DIM + gcol0;
                if (ks + KSTAGE < D_DIM)
                    __builtin_prefetch(src + (size_t)KSTAGE * H_DIM, 0, 1);
                for (int i = tid; i < KSTAGE * NCHUNK; i += 256) {
                    int k = i >> 7, n = i & (NCHUNK - 1);
                    Bls[n][k] = (bf16)src[(size_t)k * H_DIM + n];
                }
                __syncthreads();
#pragma unroll
                for (int kk = 0; kk < KSTAGE; kk += 32) {
                    const bf16* bp = &Bls[wave * 16 + lm][kk + ksel * 8];
                    v16bf b = frag16(bp, bp + 16);
#pragma unroll
                    for (int mt = 0; mt < 2; ++mt) {
                        const bf16* ap = &xs[mt * 16 + lm][ks + kk + ksel * 8];
                        v16bf a = frag16(ap, ap + 16);
                        acc[mt] = __builtin_amdgcn_wmma_f32_16x16x32_bf16(
                            false, a, false, b, (short)0, acc[mt], false,
                            false);
                    }
                }
                __syncthreads();
            }
#pragma unroll
            for (int mt = 0; mt < 2; ++mt)
#pragma unroll
                for (int r = 0; r < 8; ++r) {
                    int mrow = mt * 16 + r + (ksel << 3);
                    int col  = nc * NCHUNK + wave * 16 + lm;   // quarter-local
                    float v  = acc[mt][r] + b1g[q * QH + col];
                    hs[mrow][col] = (bf16)gelu_exact(v);
                }
        }
        __syncthreads();   // hs complete before phase 2 reads

        // ------------- Phase 2: yacc += hs @ W2[q*QH .. , :] ---------------
        for (int dc = 0; dc < D_DIM / NCHUNK; ++dc) {       // 8 chunks
            const int dcol0 = dc * NCHUNK;
            for (int ks = 0; ks < QH; ks += KSTAGE) {       // 16 stages
                const float* src =
                    W2g + (size_t)(q * QH + ks) * D_DIM + dcol0;
                if (ks + KSTAGE < QH)
                    __builtin_prefetch(src + (size_t)KSTAGE * D_DIM, 0, 1);
                for (int i = tid; i < KSTAGE * NCHUNK; i += 256) {
                    int k = i >> 7, n = i & (NCHUNK - 1);
                    Bls[n][k] = (bf16)src[(size_t)k * D_DIM + n];
                }
                __syncthreads();
#pragma unroll
                for (int kk = 0; kk < KSTAGE; kk += 32) {
                    const bf16* bp = &Bls[wave * 16 + lm][kk + ksel * 8];
                    v16bf b = frag16(bp, bp + 16);
#pragma unroll
                    for (int mt = 0; mt < 2; ++mt) {
                        const bf16* ap = &hs[mt * 16 + lm][ks + kk + ksel * 8];
                        v16bf a = frag16(ap, ap + 16);
                        yacc[dc][mt] = __builtin_amdgcn_wmma_f32_16x16x32_bf16(
                            false, a, false, b, (short)0, yacc[dc][mt], false,
                            false);
                    }
                }
                __syncthreads();
            }
        }
        __syncthreads();   // done with this hs quarter before overwrite
    }

    // ------------- Epilogue: bias, gate scale, atomic scatter --------------
#pragma unroll
    for (int dc = 0; dc < 8; ++dc)
#pragma unroll
        for (int mt = 0; mt < 2; ++mt)
#pragma unroll
            for (int r = 0; r < 8; ++r) {
                int mrow = mt * 16 + r + (ksel << 3);
                int d    = dc * NCHUNK + wave * 16 + lm;
                float wv = tw[mrow];
                if (wv != 0.0f) {
                    float v = yacc[dc][mt][r] + b2g[d];
                    atomicAdd(&out[(size_t)tk[mrow] * D_DIM + d], wv * v);
                }
            }
}

// ---------------------------------------------------------------------------
// Launch: zero output + counters, route, run grouped expert FFN.
// ---------------------------------------------------------------------------
extern "C" void kernel_launch(void* const* d_in, const int* in_sizes, int n_in,
                              void* d_out, int out_size, void* d_ws,
                              size_t ws_size, hipStream_t stream)
{
    const float* x  = (const float*)d_in[0];
    const float* Wg = (const float*)d_in[1];
    const float* bg = (const float*)d_in[2];
    const float* W1 = (const float*)d_in[3];
    const float* b1 = (const float*)d_in[4];
    const float* W2 = (const float*)d_in[5];
    const float* b2 = (const float*)d_in[6];
    float* out = (float*)d_out;

    // Workspace: [counts: E ints][tlist: E*T ints][wlist: E*T floats]
    char* ws = (char*)d_ws;
    int*   counts = (int*)ws;
    int*   tlist  = (int*)(ws + 32);
    float* wlist  = (float*)(ws + 32 + (size_t)E_EXP * T_TOK * sizeof(int));

    hipMemsetAsync(out, 0, (size_t)out_size * sizeof(float), stream);
    hipMemsetAsync(counts, 0, E_EXP * sizeof(int), stream);

    moe_router<<<T_TOK / 256, 256, 0, stream>>>(x, Wg, bg, counts, tlist,
                                                wlist);
    moe_expert_ffn<<<E_EXP * TILES, 256, 0, stream>>>(
        x, W1, b1, W2, b2, counts, tlist, wlist, out);
}
